// RBFLayer_90787018703299
// MI455X (gfx1250) — matrix operand order repaired and checked by reading, not compiled, not run on hardware
//
#include <hip/hip_runtime.h>

// RBF layer: out[b,u] = exp(-GAMMA * max(||x_b||^2 + ||c_u||^2 - 2 x_b.c_u, 0))
// x: (16384, 64) f32, centers: (4096, 64) f32, out: (16384, 4096) f32.
//
// Strategy (MI455X / gfx1250, wave32):
//  - Cross-term GEMM via V_WMMA_F32_16X16X32_BF16 with a 2-term bf16 split
//    (x = hi + lo) -> 3 WMMAs per K-step, ~fp32 accuracy, ~5 us of matrix work
//    vs the ~11 us HBM floor for the 256 MB output (23.3 TB/s).
//  - fp32 -> bf16(hi/lo) conversion staged through LDS; norms computed in fp32.
//  - Non-temporal output stores: keep x/c tiles resident in the 192 MB L2.

typedef __attribute__((ext_vector_type(16))) __bf16 v16bf;
typedef __attribute__((ext_vector_type(8)))  float  v8f;

#define RBF_GAMMA 0.01f
#define BM 128   // output rows per block (x)
#define BN 64    // output cols per block (centers)
#define KD 64    // feature dim
#define NROWS 16384
#define NCOLS 4096

__device__ __forceinline__ unsigned short f32_to_bf16_rn(float f) {
    unsigned int u = __float_as_uint(f);
    u += 0x7FFFu + ((u >> 16) & 1u);   // round-to-nearest-even
    return (unsigned short)(u >> 16);
}

union Frag {                 // one WMMA 16x16x32 bf16 operand (8 VGPRs/lane)
    v16bf v;
    unsigned int u[8];
};

__global__ __launch_bounds__(256) void rbf_wmma_bf16x3_kernel(
        const float* __restrict__ X,
        const float* __restrict__ C,
        float* __restrict__ Out) {
    __shared__ unsigned short sXh[BM * KD];   // 16 KB  x hi-plane (bf16 bits)
    __shared__ unsigned short sXl[BM * KD];   // 16 KB  x lo-plane
    __shared__ unsigned short sCh[BN * KD];   //  8 KB  c hi-plane
    __shared__ unsigned short sCl[BN * KD];   //  8 KB  c lo-plane
    __shared__ float sXsq[BM];
    __shared__ float sCsq[BN];
    __shared__ float psx[256];
    __shared__ float psc[256];

    const int tid = threadIdx.x;
    const int M0  = blockIdx.x * BM;
    const int N0  = blockIdx.y * BN;

    // ---------------- Load + split-convert X tile (128 x 64), 2 thr/row ---
    {
        const int row = tid >> 1;
        const int cb  = (tid & 1) * 32;
        const float* src = X + (size_t)(M0 + row) * KD + cb;
        float s = 0.0f;
        #pragma unroll
        for (int i = 0; i < 32; i += 4) {
            float4 f = *(const float4*)(src + i);
            float vals[4] = {f.x, f.y, f.z, f.w};
            s += f.x * f.x + f.y * f.y + f.z * f.z + f.w * f.w;
            #pragma unroll
            for (int j = 0; j < 4; ++j) {
                unsigned short h = f32_to_bf16_rn(vals[j]);
                float hf = __uint_as_float((unsigned int)h << 16);
                unsigned short l = f32_to_bf16_rn(vals[j] - hf);
                sXh[row * KD + cb + i + j] = h;
                sXl[row * KD + cb + i + j] = l;
            }
        }
        psx[tid] = s;
    }
    // ---------------- Load + split-convert C tile (64 x 64), 4 thr/row ----
    {
        const int row = tid >> 2;
        const int cb  = (tid & 3) * 16;
        const float* src = C + (size_t)(N0 + row) * KD + cb;
        float s = 0.0f;
        #pragma unroll
        for (int i = 0; i < 16; i += 4) {
            float4 f = *(const float4*)(src + i);
            float vals[4] = {f.x, f.y, f.z, f.w};
            s += f.x * f.x + f.y * f.y + f.z * f.z + f.w * f.w;
            #pragma unroll
            for (int j = 0; j < 4; ++j) {
                unsigned short h = f32_to_bf16_rn(vals[j]);
                float hf = __uint_as_float((unsigned int)h << 16);
                unsigned short l = f32_to_bf16_rn(vals[j] - hf);
                sCh[row * KD + cb + i + j] = h;
                sCl[row * KD + cb + i + j] = l;
            }
        }
        psc[tid] = s;
    }
    __syncthreads();
    if (tid < BM) {
        sXsq[tid] = psx[2 * tid] + psx[2 * tid + 1];
    } else if (tid < BM + BN) {
        const int r = tid - BM;
        sCsq[r] = psc[4 * r] + psc[4 * r + 1] + psc[4 * r + 2] + psc[4 * r + 3];
    }
    __syncthreads();

    // ---------------- WMMA compute: each wave owns a 32x32 subtile --------
    const int wave = tid >> 5;
    const int lane = tid & 31;
    const int lh   = lane >> 4;   // lane half (0/1)
    const int lm   = lane & 15;   // lane-in-half -> M row (A) / N col (B)
    const int mt   = (wave >> 1) * 32;   // 0,32,64,96
    const int nt   = (wave & 1) * 32;    // 0,32

    const v8f zero = {0.f, 0.f, 0.f, 0.f, 0.f, 0.f, 0.f, 0.f};
    v8f acc[2][2] = {{zero, zero}, {zero, zero}};

    #pragma unroll
    for (int kb = 0; kb < KD; kb += 32) {
        Frag ah[2], al[2], bh[2], bl[2];
        // A fragments (half0 K=0..15, half1 K=16..31; M = lane&15)
        #pragma unroll
        for (int mi = 0; mi < 2; ++mi) {
            const int roff = (mt + 16 * mi + lm) * KD + kb + 16 * lh;
            const unsigned short* ph = &sXh[roff];
            const unsigned short* pl = &sXl[roff];
            #pragma unroll
            for (int v = 0; v < 4; ++v) {
                ah[mi].u[v]     = *(const unsigned int*)(ph + 2 * v);
                ah[mi].u[4 + v] = *(const unsigned int*)(ph + 8 + 2 * v);
                al[mi].u[v]     = *(const unsigned int*)(pl + 2 * v);
                al[mi].u[4 + v] = *(const unsigned int*)(pl + 8 + 2 * v);
            }
        }
        // B fragments (half0 K=0..15, half1 K=16..31; N = lane&15)
        #pragma unroll
        for (int ni = 0; ni < 2; ++ni) {
            const int roff = (nt + 16 * ni + lm) * KD + kb + 16 * lh;
            const unsigned short* ph = &sCh[roff];
            const unsigned short* pl = &sCl[roff];
            #pragma unroll
            for (int v = 0; v < 8; ++v) {
                bh[ni].u[v] = *(const unsigned int*)(ph + 2 * v);
                bl[ni].u[v] = *(const unsigned int*)(pl + 2 * v);
            }
        }
        // hi*hi + hi*lo + lo*hi, fp32 accumulate (lo*lo ~2^-18, dropped)
        #pragma unroll
        for (int mi = 0; mi < 2; ++mi) {
            #pragma unroll
            for (int ni = 0; ni < 2; ++ni) {
                acc[mi][ni] = __builtin_amdgcn_wmma_f32_16x16x32_bf16(
                    false, ah[mi].v, false, bh[ni].v, (short)0, acc[mi][ni], false, false);
                acc[mi][ni] = __builtin_amdgcn_wmma_f32_16x16x32_bf16(
                    false, ah[mi].v, false, bl[ni].v, (short)0, acc[mi][ni], false, false);
                acc[mi][ni] = __builtin_amdgcn_wmma_f32_16x16x32_bf16(
                    false, al[mi].v, false, bh[ni].v, (short)0, acc[mi][ni], false, false);
            }
        }
    }

    // ---------------- Epilogue: norms + exp, non-temporal streaming store -
    #pragma unroll
    for (int mi = 0; mi < 2; ++mi) {
        #pragma unroll
        for (int ni = 0; ni < 2; ++ni) {
            const int nloc = nt + 16 * ni + lm;
            const int ncol = N0 + nloc;
            const float cs = sCsq[nloc];
            #pragma unroll
            for (int r = 0; r < 8; ++r) {
                const int mloc = mt + 16 * mi + r + 8 * lh;  // C/D layout: M = r + 8*half
                const int mrow = M0 + mloc;
                float sd = sXsq[mloc] + cs - 2.0f * acc[mi][ni][r];
                sd = fmaxf(sd, 0.0f);
                const float o = __expf(-RBF_GAMMA * sd);
                __builtin_nontemporal_store(o, Out + (size_t)mrow * NCOLS + ncol);
            }
        }
    }
}

extern "C" void kernel_launch(void* const* d_in, const int* in_sizes, int n_in,
                              void* d_out, int out_size, void* d_ws, size_t ws_size,
                              hipStream_t stream) {
    (void)in_sizes; (void)n_in; (void)d_ws; (void)ws_size; (void)out_size;
    const float* x = (const float*)d_in[0];
    const float* c = (const float*)d_in[1];
    float* out = (float*)d_out;
    dim3 grid(NROWS / BM, NCOLS / BN);   // (128, 64)
    dim3 block(256);
    rbf_wmma_bf16x3_kernel<<<grid, block, 0, stream>>>(x, c, out);
}
